// BigBirdClassifier_57586921505277
// MI455X (gfx1250) — compile-verified, matrix operand order checked by reading, not run on hardware
//
#include <hip/hip_runtime.h>
#include <hip/hip_fp16.h>

typedef __attribute__((ext_vector_type(16))) _Float16 v16h;
typedef __attribute__((ext_vector_type(8)))  _Float16 v8h;
typedef __attribute__((ext_vector_type(8)))  float    v8f;

union F16Frag { v16h v; v8h h[2]; };

constexpr int Bx = 4, S = 1920, D = 1024, H = 8, L = 6;
constexpr int NB = 30, NI = 28, R = 3, FF = 4096, DH = 128, NLBL = 1000;
constexpr int BS = Bx * S; // 7680
constexpr float NEGV = -1.0e9f;

// ---------------------------------------------------------------------------
// Transpose + convert: out[n][k] = (f16) in[k][n].  K, N multiples of 32.
// Block 256 thr = 32x8, 32x32 tiles via LDS.
// ---------------------------------------------------------------------------
__global__ __launch_bounds__(256) void transpose_cvt(
    const float* __restrict__ in, _Float16* __restrict__ out, int K, int N)
{
  __shared__ float tile[32][33];
  const int k0 = blockIdx.y * 32, n0 = blockIdx.x * 32;
  const int tx = threadIdx.x & 31, ty = threadIdx.x >> 5;
#pragma unroll
  for (int i = 0; i < 4; ++i)
    tile[ty + i * 8][tx] = in[(size_t)(k0 + ty + i * 8) * N + n0 + tx];
  __syncthreads();
#pragma unroll
  for (int i = 0; i < 4; ++i)
    out[(size_t)(n0 + ty + i * 8) * K + k0 + tx] = (_Float16)tile[tx][ty + i * 8];
}

// ---------------------------------------------------------------------------
// WMMA GEMM: C[M,N] = A[M,K](f16) @ Bt[N,K](f16, pre-transposed) + bias
// EPI 0: f32 out.  EPI 1: gelu -> f16 out.  EPI 2: qkv scatter -> f16 [B,H,S,DH]
// Block = 256 threads (8 waves), 128x128 tile, K-step 32, double-buffered LDS.
// ---------------------------------------------------------------------------
template <int EPI>
__global__ __launch_bounds__(256) void gemm_wmma(
    const _Float16* __restrict__ A, const _Float16* __restrict__ Bt,
    const float* __restrict__ bias, float* __restrict__ out32,
    _Float16* __restrict__ out16, int M, int N, int K)
{
  __shared__ __align__(16) _Float16 As[2][128][40];
  __shared__ __align__(16) _Float16 Bs[2][128][40];

  const int t  = threadIdx.x;
  const int bm = blockIdx.y * 128, bn = blockIdx.x * 128;
  const int lane = t & 31, wv = t >> 5;
  const int wr = wv & 3, wc = wv >> 2;
  const int ln15 = lane & 15;
  const int kbo  = (lane & 16) ? 8 : 0;
  const int half8 = (lane & 16) ? 8 : 0;

  v8f acc[2][4] = {};

  const int row = t >> 1, part = (t & 1) * 16;
  const _Float16* aptr = A  + (size_t)(bm + row) * K + part;
  const _Float16* bptr = Bt + (size_t)(bn + row) * K + part;

  // prologue: tile 0 -> buffer 0
  {
    const v8h* ga = (const v8h*)aptr;
    *(v8h*)&As[0][row][part]     = ga[0];
    *(v8h*)&As[0][row][part + 8] = ga[1];
    const v8h* gb = (const v8h*)bptr;
    *(v8h*)&Bs[0][row][part]     = gb[0];
    *(v8h*)&Bs[0][row][part + 8] = gb[1];
  }
  __syncthreads();

  const int nk = K >> 5;
  for (int ki = 0; ki < nk; ++ki) {
    const int buf = ki & 1;
    if (ki + 1 < nk) {
      const int k1 = (ki + 1) * 32;
      const v8h* ga = (const v8h*)(aptr + k1);
      *(v8h*)&As[buf ^ 1][row][part]     = ga[0];
      *(v8h*)&As[buf ^ 1][row][part + 8] = ga[1];
      const v8h* gb = (const v8h*)(bptr + k1);
      *(v8h*)&Bs[buf ^ 1][row][part]     = gb[0];
      *(v8h*)&Bs[buf ^ 1][row][part + 8] = gb[1];
    }

    F16Frag af[2], bf[4];
#pragma unroll
    for (int mt = 0; mt < 2; ++mt) {
      const _Float16* p = &As[buf][wr * 32 + mt * 16 + ln15][kbo];
      af[mt].h[0] = *(const v8h*)p;
      af[mt].h[1] = *(const v8h*)(p + 16);
    }
#pragma unroll
    for (int nt = 0; nt < 4; ++nt) {
      const _Float16* p = &Bs[buf][wc * 64 + nt * 16 + ln15][kbo];
      bf[nt].h[0] = *(const v8h*)p;
      bf[nt].h[1] = *(const v8h*)(p + 16);
    }
#pragma unroll
    for (int mt = 0; mt < 2; ++mt)
#pragma unroll
      for (int nt = 0; nt < 4; ++nt)
        acc[mt][nt] = __builtin_amdgcn_wmma_f32_16x16x32_f16(
            false, af[mt].v, false, bf[nt].v, (short)0, acc[mt][nt], false, false);
    __syncthreads();
  }

#pragma unroll
  for (int mt = 0; mt < 2; ++mt) {
#pragma unroll
    for (int nt = 0; nt < 4; ++nt) {
      const int n = bn + wc * 64 + nt * 16 + ln15;
      const float bsv = bias[n];
#pragma unroll
      for (int r = 0; r < 8; ++r) {
        const int m = bm + wr * 32 + mt * 16 + r + half8;
        const float val = acc[mt][nt][r] + bsv;
        if constexpr (EPI == 0) {
          out32[(size_t)m * N + n] = val;
        } else if constexpr (EPI == 1) {
          const float x = val;
          const float g = 0.5f * x *
              (1.0f + tanhf(0.7978845608028654f * (x + 0.044715f * x * x * x)));
          out16[(size_t)m * N + n] = (_Float16)g;
        } else {
          const int hh = n >> 7, dd = n & 127;
          const int bb = m / S, ss = m % S;
          out16[(((size_t)bb * H + hh) * S + ss) * DH + dd] = (_Float16)val;
        }
      }
    }
  }
}

// ---------------------------------------------------------------------------
// Residual + LayerNorm: y = LN(A[row] + Badd[row % bmod]) * gamma + beta
// Writes f32 and f16 copies. One block (256 thr) per row of D=1024.
// ---------------------------------------------------------------------------
__global__ __launch_bounds__(256) void ln_residual(
    const float* __restrict__ A, const float* __restrict__ Badd, int bmod,
    const float* __restrict__ gamma, const float* __restrict__ beta,
    float* __restrict__ out32, _Float16* __restrict__ out16)
{
  __shared__ float red[256];
  const int row = blockIdx.x, t = threadIdx.x;
  const float* a  = A + (size_t)row * D;
  const float* bb = Badd + (size_t)(row % bmod) * D;

  float x[4]; float s = 0.f;
#pragma unroll
  for (int i = 0; i < 4; ++i) { x[i] = a[t + i * 256] + bb[t + i * 256]; s += x[i]; }
  red[t] = s; __syncthreads();
  for (int off = 128; off > 0; off >>= 1) { if (t < off) red[t] += red[t + off]; __syncthreads(); }
  const float mean = red[0] * (1.0f / D);
  __syncthreads();
  float v = 0.f;
#pragma unroll
  for (int i = 0; i < 4; ++i) { const float d = x[i] - mean; v += d * d; }
  red[t] = v; __syncthreads();
  for (int off = 128; off > 0; off >>= 1) { if (t < off) red[t] += red[t + off]; __syncthreads(); }
  const float rstd = rsqrtf(red[0] * (1.0f / D) + 1e-12f);
#pragma unroll
  for (int i = 0; i < 4; ++i) {
    const int c = t + i * 256;
    const float y = (x[i] - mean) * rstd * gamma[c] + beta[c];
    out32[(size_t)row * D + c] = y;
    out16[(size_t)row * D + c] = (_Float16)y;
  }
}

// ---------------------------------------------------------------------------
// BigBird attention, flash-style online softmax with WMMA.
// Q/K/V f16 in [B,H,S,DH]. Block = 128 thr (4 waves), grid (NB, H, B).
// Wave w owns query rows w*16..w*16+15 of this 64-row query block.
// ---------------------------------------------------------------------------
__global__ __launch_bounds__(128) void bigbird_attn(
    const _Float16* __restrict__ Q, const _Float16* __restrict__ Kt,
    const _Float16* __restrict__ V, const float* __restrict__ mask,
    const int* __restrict__ randb, _Float16* __restrict__ O)
{
  __shared__ __align__(16) _Float16 Ks[64][136];
  __shared__ __align__(16) _Float16 Vt[128][72];
  __shared__ __align__(16) _Float16 Ps[64][72];
  __shared__ float ms[64];

  const int qb = blockIdx.x, hh = blockIdx.y, b = blockIdx.z;
  const int t = threadIdx.x, lane = t & 31, wv = t >> 5;
  const int m0 = wv * 16;
  const int ln15 = lane & 15;
  const int kbo = (lane & 16) ? 8 : 0;
  const int half8 = (lane & 16) ? 8 : 0;
  const float scale = 0.08838834764831845f; // 1/sqrt(128)

  const size_t headbase = ((size_t)b * H + hh) * S;

  // chunk list (block-uniform)
  int kblocks[8]; float kkeep[8]; int nch;
  const bool isglob = (qb == 0) || (qb == NB - 1);
  if (isglob) {
    nch = NB;
  } else {
    const int i = qb - 1;
    const int* rb = randb + (hh * NI + i) * R;
    kblocks[0] = 0; kblocks[1] = NB - 1;
    kblocks[2] = i - 1; kblocks[3] = i; kblocks[4] = i + 1;
    kblocks[5] = rb[0]; kblocks[6] = rb[1]; kblocks[7] = rb[2];
    nch = 8;
#pragma unroll
    for (int j = 0; j < 8; ++j) {
      float kp = 1.f;
      for (int j2 = 0; j2 < j; ++j2) if (kblocks[j2] == kblocks[j]) kp = 0.f;
      kkeep[j] = kp;
    }
  }

  // Q fragments for this wave's 16 rows (K-dim = 128 -> 4 steps)
  F16Frag qf[4];
  {
    const _Float16* qp = Q + (headbase + (size_t)qb * 64 + m0 + ln15) * DH + kbo;
#pragma unroll
    for (int kt = 0; kt < 4; ++kt) {
      qf[kt].h[0] = *(const v8h*)(qp + kt * 32);
      qf[kt].h[1] = *(const v8h*)(qp + kt * 32 + 16);
    }
  }

  v8f oacc[8] = {};
  float rm[8], rs[8];
#pragma unroll
  for (int r = 0; r < 8; ++r) { rm[r] = -3.0e38f; rs[r] = 0.f; }

  for (int c = 0; c < nch; ++c) {
    const int blk = isglob ? c : kblocks[c];
    if (!isglob && kkeep[c] == 0.f) continue; // uniform across block
    const int s0 = blk * 64;

    __syncthreads();
    {
      const int row = t >> 1, c0 = (t & 1) * 64;
      const _Float16* kp = Kt + (headbase + s0 + row) * DH + c0;
#pragma unroll
      for (int j = 0; j < 8; ++j)
        *(v8h*)&Ks[row][c0 + j * 8] = *(const v8h*)(kp + j * 8);
      const _Float16* vp = V + (headbase + s0 + row) * DH + c0;
#pragma unroll
      for (int j = 0; j < 8; ++j) {
        v8h vvv = *(const v8h*)(vp + j * 8);
#pragma unroll
        for (int e = 0; e < 8; ++e) Vt[c0 + j * 8 + e][row] = vvv[e];
      }
      if (t < 64) ms[t] = mask[(size_t)b * S + s0 + t];
    }
    __syncthreads();

    // scores: 16 rows x 64 keys per wave
    v8f sacc[4] = {};
#pragma unroll
    for (int nt = 0; nt < 4; ++nt) {
      const _Float16* bp = &Ks[nt * 16 + ln15][kbo];
#pragma unroll
      for (int kt = 0; kt < 4; ++kt) {
        F16Frag bfr;
        bfr.h[0] = *(const v8h*)(bp + kt * 32);
        bfr.h[1] = *(const v8h*)(bp + kt * 32 + 16);
        sacc[nt] = __builtin_amdgcn_wmma_f32_16x16x32_f16(
            false, qf[kt].v, false, bfr.v, (short)0, sacc[nt], false, false);
      }
    }

    float vals[4][8];
#pragma unroll
    for (int nt = 0; nt < 4; ++nt) {
      const float mval = ms[nt * 16 + ln15];
#pragma unroll
      for (int r = 0; r < 8; ++r)
        vals[nt][r] = (mval > 0.f) ? sacc[nt][r] * scale : NEGV;
    }

    float corr[8];
#pragma unroll
    for (int r = 0; r < 8; ++r) {
      float m = fmaxf(fmaxf(vals[0][r], vals[1][r]), fmaxf(vals[2][r], vals[3][r]));
      m = fmaxf(m, __shfl_xor(m, 1));
      m = fmaxf(m, __shfl_xor(m, 2));
      m = fmaxf(m, __shfl_xor(m, 4));
      m = fmaxf(m, __shfl_xor(m, 8));
      const float nm = fmaxf(rm[r], m);
      corr[r] = expf(rm[r] - nm);
      rm[r] = nm;
    }

    float psum[8] = {};
#pragma unroll
    for (int nt = 0; nt < 4; ++nt) {
#pragma unroll
      for (int r = 0; r < 8; ++r) {
        const float p = expf(vals[nt][r] - rm[r]);
        psum[r] += p;
        Ps[m0 + r + half8][nt * 16 + ln15] = (_Float16)p;
      }
    }
#pragma unroll
    for (int r = 0; r < 8; ++r) {
      float sm = psum[r];
      sm += __shfl_xor(sm, 1); sm += __shfl_xor(sm, 2);
      sm += __shfl_xor(sm, 4); sm += __shfl_xor(sm, 8);
      rs[r] = rs[r] * corr[r] + sm;
    }
#pragma unroll
    for (int nt = 0; nt < 8; ++nt)
#pragma unroll
      for (int r = 0; r < 8; ++r)
        oacc[nt][r] *= corr[r];
    __syncthreads();

    // O += P(16x64) @ V(64x128)
#pragma unroll
    for (int kt2 = 0; kt2 < 2; ++kt2) {
      F16Frag pf;
      const _Float16* pp = &Ps[m0 + ln15][kt2 * 32 + kbo];
      pf.h[0] = *(const v8h*)pp;
      pf.h[1] = *(const v8h*)(pp + 16);
#pragma unroll
      for (int nt = 0; nt < 8; ++nt) {
        F16Frag vf;
        const _Float16* vpp = &Vt[nt * 16 + ln15][kt2 * 32 + kbo];
        vf.h[0] = *(const v8h*)vpp;
        vf.h[1] = *(const v8h*)(vpp + 16);
        oacc[nt] = __builtin_amdgcn_wmma_f32_16x16x32_f16(
            false, pf.v, false, vf.v, (short)0, oacc[nt], false, false);
      }
    }
  }

  float inv[8];
#pragma unroll
  for (int r = 0; r < 8; ++r) inv[r] = 1.0f / rs[r];
#pragma unroll
  for (int nt = 0; nt < 8; ++nt) {
    const int n = hh * DH + nt * 16 + ln15;
#pragma unroll
    for (int r = 0; r < 8; ++r) {
      const int srow = qb * 64 + m0 + r + half8;
      O[((size_t)b * S + srow) * D + n] = (_Float16)(oacc[nt][r] * inv[r]);
    }
  }
}

// ---------------------------------------------------------------------------
// Classifier head (tiny): relu(cls @ Wc1 + bc1) @ Wc2 + bc2
// ---------------------------------------------------------------------------
__global__ void cls1_kernel(const float* __restrict__ h32, const float* __restrict__ Wc1,
                            const float* __restrict__ bc1, float* __restrict__ cmid)
{
  const int o = blockIdx.x * blockDim.x + threadIdx.x;
  if (o >= Bx * 512) return;
  const int bb = o / 512, j = o % 512;
  const float* row = h32 + (size_t)bb * S * D; // h[b, 0, :]
  float acc = bc1[j];
  for (int k = 0; k < D; ++k) acc += row[k] * Wc1[(size_t)k * 512 + j];
  cmid[o] = fmaxf(acc, 0.f);
}

__global__ void cls2_kernel(const float* __restrict__ cmid, const float* __restrict__ Wc2,
                            const float* __restrict__ bc2, float* __restrict__ out)
{
  const int o = blockIdx.x * blockDim.x + threadIdx.x;
  if (o >= Bx * NLBL) return;
  const int bb = o / NLBL, j = o % NLBL;
  float acc = bc2[j];
  for (int k = 0; k < 512; ++k) acc += cmid[bb * 512 + k] * Wc2[(size_t)k * NLBL + j];
  out[o] = acc;
}

// ---------------------------------------------------------------------------
extern "C" void kernel_launch(void* const* d_in, const int* in_sizes, int n_in,
                              void* d_out, int out_size, void* d_ws, size_t ws_size,
                              hipStream_t stream)
{
  (void)in_sizes; (void)n_in; (void)out_size; (void)ws_size;
  const float* x     = (const float*)d_in[0];
  const float* amask = (const float*)d_in[1];
  const float* pos   = (const float*)d_in[2];
  const float* embln = (const float*)d_in[3];
  const float* Wq    = (const float*)d_in[4];
  const float* bq    = (const float*)d_in[5];
  const float* Wk    = (const float*)d_in[6];
  const float* bk    = (const float*)d_in[7];
  const float* Wv    = (const float*)d_in[8];
  const float* bv    = (const float*)d_in[9];
  const float* Wo    = (const float*)d_in[10];
  const float* bo    = (const float*)d_in[11];
  const float* ln1   = (const float*)d_in[12];
  const float* W1    = (const float*)d_in[13];
  const float* b1    = (const float*)d_in[14];
  const float* W2    = (const float*)d_in[15];
  const float* b2    = (const float*)d_in[16];
  const float* ln2   = (const float*)d_in[17];
  const float* Wc1   = (const float*)d_in[18];
  const float* bc1   = (const float*)d_in[19];
  const float* Wc2   = (const float*)d_in[20];
  const float* bc2   = (const float*)d_in[21];
  const int*   rnd   = (const int*)d_in[22];

  char* ws = (char*)d_ws;
  size_t off = 0;
  auto alloc = [&](size_t bytes) -> void* {
    void* p = ws + off;
    off += (bytes + 255) & ~(size_t)255;
    return p;
  };
  float*    h32   = (float*)alloc((size_t)BS * D * 4);
  float*    tmp32 = (float*)alloc((size_t)BS * D * 4);
  _Float16* h16   = (_Float16*)alloc((size_t)BS * D * 2);
  _Float16* q16   = (_Float16*)alloc((size_t)BS * D * 2);
  _Float16* k16   = (_Float16*)alloc((size_t)BS * D * 2);
  _Float16* v16   = (_Float16*)alloc((size_t)BS * D * 2);
  _Float16* ao16  = (_Float16*)alloc((size_t)BS * D * 2);
  _Float16* mid16 = (_Float16*)alloc((size_t)BS * FF * 2);
  float*    cmid  = (float*)alloc((size_t)Bx * 512 * 4);
  // per-layer transposed f16 weights (reused across layers)
  _Float16* wtq = (_Float16*)alloc((size_t)D * D * 2);
  _Float16* wtk = (_Float16*)alloc((size_t)D * D * 2);
  _Float16* wtv = (_Float16*)alloc((size_t)D * D * 2);
  _Float16* wto = (_Float16*)alloc((size_t)D * D * 2);
  _Float16* wt1 = (_Float16*)alloc((size_t)D * FF * 2); // [FF][D]
  _Float16* wt2 = (_Float16*)alloc((size_t)FF * D * 2); // [D][FF]

  const dim3 blk256(256);
  const dim3 grid_d(D / 128, BS / 128);    // N=1024 GEMMs
  const dim3 grid_ff(FF / 128, BS / 128);  // N=4096 GEMM
  const dim3 tgrid_dd(D / 32, D / 32);     // W[D][D] transpose
  const dim3 tgrid_w1(FF / 32, D / 32);    // W1[D][FF] -> [FF][D]
  const dim3 tgrid_w2(D / 32, FF / 32);    // W2[FF][D] -> [D][FF]

  // embed: h = LN(x + pos_emb)
  ln_residual<<<dim3(BS), blk256, 0, stream>>>(x, pos, S, embln, embln + D, h32, h16);

  for (int l = 0; l < L; ++l) {
    const float* Wql = Wq + (size_t)l * D * D;  const float* bql = bq + (size_t)l * D;
    const float* Wkl = Wk + (size_t)l * D * D;  const float* bkl = bk + (size_t)l * D;
    const float* Wvl = Wv + (size_t)l * D * D;  const float* bvl = bv + (size_t)l * D;
    const float* Wol = Wo + (size_t)l * D * D;  const float* bol = bo + (size_t)l * D;
    const float* W1l = W1 + (size_t)l * D * FF; const float* b1l = b1 + (size_t)l * FF;
    const float* W2l = W2 + (size_t)l * FF * D; const float* b2l = b2 + (size_t)l * D;
    const float* g1 = ln1 + (size_t)l * 2 * D;
    const float* g2 = ln2 + (size_t)l * 2 * D;

    // convert this layer's weights to transposed f16
    transpose_cvt<<<tgrid_dd, blk256, 0, stream>>>(Wql, wtq, D, D);
    transpose_cvt<<<tgrid_dd, blk256, 0, stream>>>(Wkl, wtk, D, D);
    transpose_cvt<<<tgrid_dd, blk256, 0, stream>>>(Wvl, wtv, D, D);
    transpose_cvt<<<tgrid_dd, blk256, 0, stream>>>(Wol, wto, D, D);
    transpose_cvt<<<tgrid_w1, blk256, 0, stream>>>(W1l, wt1, D, FF);
    transpose_cvt<<<tgrid_w2, blk256, 0, stream>>>(W2l, wt2, FF, D);

    gemm_wmma<2><<<grid_d, blk256, 0, stream>>>(h16, wtq, bql, nullptr, q16, BS, D, D);
    gemm_wmma<2><<<grid_d, blk256, 0, stream>>>(h16, wtk, bkl, nullptr, k16, BS, D, D);
    gemm_wmma<2><<<grid_d, blk256, 0, stream>>>(h16, wtv, bvl, nullptr, v16, BS, D, D);

    bigbird_attn<<<dim3(NB, H, Bx), dim3(128), 0, stream>>>(q16, k16, v16, amask, rnd, ao16);

    gemm_wmma<0><<<grid_d, blk256, 0, stream>>>(ao16, wto, bol, tmp32, nullptr, BS, D, D);
    ln_residual<<<dim3(BS), blk256, 0, stream>>>(h32, tmp32, BS, g1, g1 + D, h32, h16);

    gemm_wmma<1><<<grid_ff, blk256, 0, stream>>>(h16, wt1, b1l, nullptr, mid16, BS, FF, D);
    gemm_wmma<0><<<grid_d, blk256, 0, stream>>>(mid16, wt2, b2l, tmp32, nullptr, BS, D, FF);
    ln_residual<<<dim3(BS), blk256, 0, stream>>>(h32, tmp32, BS, g2, g2 + D, h32, h16);
  }

  cls1_kernel<<<dim3((Bx * 512 + 255) / 256), blk256, 0, stream>>>(h32, Wc1, bc1, cmid);
  cls2_kernel<<<dim3((Bx * NLBL + 255) / 256), blk256, 0, stream>>>(cmid, Wc2, bc2, (float*)d_out);
}
